// MixAwareLoss_43061342109810
// MI455X (gfx1250) — compile-verified, matrix operand order checked by reading, not run on hardware
//
#include <hip/hip_runtime.h>
#include <hip/hip_bf16.h>
#include <math.h>

// Problem constants (from reference)
#define BB 4096
#define DD 256
#define EPS_F 1e-8f
#define INV_TNEG 10.0f   // 1 / 0.1
#define INV_TPOS 20.0f   // 1 / 0.05

// GEMM staging geometry
#define KC    32                 // K elements per stage
#define PITCH 40                 // padded LDS row pitch (bf16 elems) -> conflict-free
#define TILE_E (128 * PITCH)     // elements per (array, buffer)
#define SMEM_BYTES (2 * 4 * TILE_E * 2)   // 2 bufs x 4 arrays x bf16 = 81920 B

typedef __bf16 bf16_t;
typedef bf16_t v16bf __attribute__((ext_vector_type(16)));
typedef float  v8f   __attribute__((ext_vector_type(8)));

union FragU {
    struct { uint4 a, b; } u;   // two 16-byte chunks = 8 VGPRs
    v16bf v;
};

// CDNA5 16-bit A-fragment layout (16x32 MxK) == B-fragment layout (32x16 KxN)
// for a Gram matrix loaded from the same row-major array:
//   lanes 0-15 : row = row16+lane,    K = 0..7  and 16..23 of the stage
//   lanes 16-31: row = row16+lane-16, K = 8..15 and 24..31 of the stage
__device__ __forceinline__ v16bf load_frag_lds(const bf16_t* __restrict__ tile,
                                               int row16, int lane) {
    const int r  = row16 + (lane & 15);
    const int kk = (lane & 16) ? 8 : 0;
    const uint4* p = reinterpret_cast<const uint4*>(tile + r * PITCH + kk);
    FragU f;
    f.u.a = p[0];   // K = kk .. kk+7
    f.u.b = p[2];   // K = kk+16 .. kk+23
    return f.v;
}

struct alignas(8) BF4 { bf16_t x, y, z, w; };

__device__ __forceinline__ void split_store(bf16_t* __restrict__ hi,
                                            bf16_t* __restrict__ lo,
                                            size_t base, float4 v) {
    BF4 h, l;
    h.x = (bf16_t)v.x; l.x = (bf16_t)(v.x - (float)h.x);
    h.y = (bf16_t)v.y; l.y = (bf16_t)(v.y - (float)h.y);
    h.z = (bf16_t)v.z; l.z = (bf16_t)(v.z - (float)h.z);
    h.w = (bf16_t)v.w; l.w = (bf16_t)(v.w - (float)h.w);
    *reinterpret_cast<BF4*>(hi + base) = h;
    *reinterpret_cast<BF4*>(lo + base) = l;
}

// ---------------------------------------------------------------------------
// Kernel 1: per-row L2 norms (all 2B rows) + compensated bf16 split of X[:B]
// ---------------------------------------------------------------------------
__global__ __launch_bounds__(256)
void prep_kernel(const float* __restrict__ X,
                 bf16_t* __restrict__ Xhi, bf16_t* __restrict__ Xlo,
                 float* __restrict__ norms) {
    const int row  = blockIdx.x * 8 + (threadIdx.x >> 5);
    const int lane = threadIdx.x & 31;
    if (row >= 2 * BB) return;

    const float4* r4 = reinterpret_cast<const float4*>(X + (size_t)row * DD);
    float4 a = r4[lane];
    float4 b = r4[lane + 32];

    float ss = a.x*a.x + a.y*a.y + a.z*a.z + a.w*a.w
             + b.x*b.x + b.y*b.y + b.z*b.z + b.w*b.w;
    #pragma unroll
    for (int m = 1; m < 32; m <<= 1) ss += __shfl_xor(ss, m, 32);
    if (lane == 0) norms[row] = sqrtf(ss);

    if (row < BB) {
        const size_t base = (size_t)row * DD;
        split_store(Xhi, Xlo, base + lane * 4,        a);
        split_store(Xhi, Xlo, base + (lane + 32) * 4, b);
    }
}

// ---------------------------------------------------------------------------
// Kernel 2: positives[i] = dot(x_i, x_{i+B}) / max(n_i * n_{i+B}, eps)
// ---------------------------------------------------------------------------
__global__ __launch_bounds__(256)
void positives_kernel(const float* __restrict__ X,
                      const float* __restrict__ norms,
                      float* __restrict__ possim) {
    const int i    = blockIdx.x * 8 + (threadIdx.x >> 5);
    const int lane = threadIdx.x & 31;
    if (i >= BB) return;

    const float4* ra = reinterpret_cast<const float4*>(X + (size_t)i * DD);
    const float4* rb = reinterpret_cast<const float4*>(X + (size_t)(i + BB) * DD);
    float4 a1 = ra[lane], a2 = ra[lane + 32];
    float4 b1 = rb[lane], b2 = rb[lane + 32];

    float d = a1.x*b1.x + a1.y*b1.y + a1.z*b1.z + a1.w*b1.w
            + a2.x*b2.x + a2.y*b2.y + a2.z*b2.z + a2.w*b2.w;
    #pragma unroll
    for (int m = 1; m < 32; m <<= 1) d += __shfl_xor(d, m, 32);
    if (lane == 0)
        possim[i] = d / fmaxf(norms[i] * norms[i + BB], EPS_F);
}

// ---------------------------------------------------------------------------
// Async staging: one wave stages half of one of the 4 arrays per stage.
//   w>>1 selects array (0:Ahi 1:Alo 2:Bhi 3:Blo), w&1 selects 64-row half.
//   Each of the 8 async b128 instructions moves 32 lanes x 16B = 512 B
//   (8 rows x 4 chunks); lane l -> row (l>>2), chunk (l&3).
// Tracked by ASYNCcnt; in-order completion lets us use partial waits.
// ---------------------------------------------------------------------------
__device__ __forceinline__ void issue_stage(char* smem, int buf,
        const bf16_t* __restrict__ Xhi, const bf16_t* __restrict__ Xlo,
        int rowTile, int colTile, int k0, int w, int lane) {
    const int arr  = w >> 1;                 // 0..3
    const int half = w & 1;                  // which 64-row half
    const bf16_t* src = (arr & 1) ? Xlo : Xhi;
    const int tb = (arr & 2) ? colTile : rowTile;
    const int r0 = half * 64 + (lane >> 2);  // this lane's row for i = 0
    const int kc = (lane & 3) * 8;           // this lane's 8-elem chunk

    const bf16_t* g = src + (size_t)(tb + r0) * DD + k0 + kc;
    bf16_t* dstArr  = (bf16_t*)smem + (size_t)(buf * 4 + arr) * TILE_E;
    unsigned int lb = (unsigned int)(size_t)(dstArr + r0 * PITCH + kc);

    #pragma unroll
    for (int i = 0; i < 8; ++i) {
        asm volatile("global_load_async_to_lds_b128 %0, %1, off"
                     :: "v"(lb), "v"(g) : "memory");
        g  += 8 * DD;                          // +8 rows in global
        lb += 8 * PITCH * (unsigned)sizeof(bf16_t);  // +8 rows in LDS
    }
}

// ---------------------------------------------------------------------------
// Kernel 3: Gram GEMM (compensated bf16 WMMA) + exp row-sum partials.
// 32x32 blocks; 128x128 tile per block; 8 waves arranged 4(rows) x 2(cols),
// each wave owns a 32x64 subtile = 2x4 fragments; 3 bf16 WMMAs per fragment
// per 32-wide K stage (hi*hi + hi*lo + lo*hi). Operands double-buffered in
// LDS via async global->LDS DMA.
// ---------------------------------------------------------------------------
__global__ __launch_bounds__(256)
void gram_exp_kernel(const bf16_t* __restrict__ Xhi,
                     const bf16_t* __restrict__ Xlo,
                     const float* __restrict__ norms,
                     float* __restrict__ partial) {
    extern __shared__ char smem[];

    const int lane = threadIdx.x & 31;
    const int w    = threadIdx.x >> 5;
    const int wr   = w >> 1;          // 0..3
    const int wc   = w & 1;           // 0..1
    const int rowTile = blockIdx.y * 128;
    const int colTile = blockIdx.x * 128;
    const int rowBase = rowTile + wr * 32;
    const int colBase = colTile + wc * 64;

    const v8f zero = {0.f, 0.f, 0.f, 0.f, 0.f, 0.f, 0.f, 0.f};
    v8f acc[2][4];
    #pragma unroll
    for (int mi = 0; mi < 2; ++mi)
        #pragma unroll
        for (int ni = 0; ni < 4; ++ni) acc[mi][ni] = zero;

    // Prologue: start stage 0.
    issue_stage(smem, 0, Xhi, Xlo, rowTile, colTile, 0, w, lane);

    #pragma unroll
    for (int s = 0; s < DD / KC; ++s) {               // 8 stages
        if (s < DD / KC - 1) {
            issue_stage(smem, (s + 1) & 1, Xhi, Xlo, rowTile, colTile,
                        (s + 1) * KC, w, lane);
            // 16 outstanding; in-order => <=8 means stage s is complete.
            asm volatile("s_wait_asynccnt 0x8" ::: "memory");
        } else {
            asm volatile("s_wait_asynccnt 0x0" ::: "memory");
        }
        __syncthreads();   // stage s visible to all waves

        const bf16_t* base = (const bf16_t*)smem + (size_t)((s & 1) * 4) * TILE_E;
        const bf16_t* Ah = base;
        const bf16_t* Al = base + TILE_E;
        const bf16_t* Bh = base + 2 * TILE_E;
        const bf16_t* Bl = base + 3 * TILE_E;

        v16bf ah[2], al[2], bh[4], bl[4];
        #pragma unroll
        for (int mi = 0; mi < 2; ++mi) {
            ah[mi] = load_frag_lds(Ah, wr * 32 + mi * 16, lane);
            al[mi] = load_frag_lds(Al, wr * 32 + mi * 16, lane);
        }
        #pragma unroll
        for (int ni = 0; ni < 4; ++ni) {
            bh[ni] = load_frag_lds(Bh, wc * 64 + ni * 16, lane);
            bl[ni] = load_frag_lds(Bl, wc * 64 + ni * 16, lane);
        }
        #pragma unroll
        for (int mi = 0; mi < 2; ++mi)
            #pragma unroll
            for (int ni = 0; ni < 4; ++ni) {
                acc[mi][ni] = __builtin_amdgcn_wmma_f32_16x16x32_bf16(
                    false, ah[mi], false, bh[ni], (short)0, acc[mi][ni], false, false);
                acc[mi][ni] = __builtin_amdgcn_wmma_f32_16x16x32_bf16(
                    false, ah[mi], false, bl[ni], (short)0, acc[mi][ni], false, false);
                acc[mi][ni] = __builtin_amdgcn_wmma_f32_16x16x32_bf16(
                    false, al[mi], false, bh[ni], (short)0, acc[mi][ni], false, false);
            }

        __syncthreads();   // all waves done reading buffer parity s&1
    }

    // ---- epilogue ----
    const int half = lane >> 4;    // 0: rows m0..m0+7 ; 1: rows m0+8..m0+15
    const int ln   = lane & 15;
    const int chunk = blockIdx.x * 2 + wc;          // 0..63
    float* __restrict__ pchunk = partial + (size_t)chunk * BB;

    float ncol[4];
    #pragma unroll
    for (int ni = 0; ni < 4; ++ni)
        ncol[ni] = norms[colBase + ni * 16 + ln];

    #pragma unroll
    for (int mi = 0; mi < 2; ++mi) {
        const float nrowLane = norms[rowBase + mi * 16 + ln];
        #pragma unroll
        for (int v = 0; v < 8; ++v) {
            const int i  = rowBase + mi * 16 + half * 8 + v;   // global row
            const float n_i = __shfl(nrowLane, half * 8 + v, 32);
            float racc = 0.f;
            #pragma unroll
            for (int ni = 0; ni < 4; ++ni) {
                const int j = colBase + ni * 16 + ln;           // global col
                const float sim = acc[mi][ni][v] / fmaxf(n_i * ncol[ni], EPS_F);
                racc += (i == j) ? 0.f : __expf(sim * INV_TNEG);
            }
            racc += __shfl_xor(racc, 1, 32);
            racc += __shfl_xor(racc, 2, 32);
            racc += __shfl_xor(racc, 4, 32);
            racc += __shfl_xor(racc, 8, 32);
            if (ln == 0) pchunk[i] = racc;   // lane 0 / lane 16 -> distinct rows
        }
    }
}

// ---------------------------------------------------------------------------
// Kernel 4: loss = mean_i( log(sum_chunks partial[c][i]) - pos[i]/T_POS )
// Single block, fixed-order reduction -> fully deterministic.
// ---------------------------------------------------------------------------
__global__ __launch_bounds__(256)
void finalize_kernel(const float* __restrict__ partial,
                     const float* __restrict__ possim,
                     float* __restrict__ out) {
    __shared__ float red[256];
    const int t = threadIdx.x;
    float local = 0.f;
    for (int r = t; r < BB; r += 256) {
        float s = 0.f;
        #pragma unroll 8
        for (int c = 0; c < 64; ++c) s += partial[(size_t)c * BB + r];
        local += logf(s) - possim[r] * INV_TPOS;
    }
    red[t] = local;
    __syncthreads();
    #pragma unroll
    for (int off = 128; off > 0; off >>= 1) {
        if (t < off) red[t] += red[t + off];
        __syncthreads();
    }
    if (t == 0) out[0] = red[0] / (float)BB;
}

// ---------------------------------------------------------------------------
extern "C" void kernel_launch(void* const* d_in, const int* in_sizes, int n_in,
                              void* d_out, int out_size, void* d_ws, size_t ws_size,
                              hipStream_t stream) {
    const float* X  = (const float*)d_in[0];
    float* out      = (float*)d_out;
    char* ws        = (char*)d_ws;

    const size_t XB    = (size_t)BB * DD * sizeof(bf16_t);   // 2 MB each
    const size_t NORMB = (size_t)(2 * BB) * sizeof(float);   // 32 KB
    const size_t POSB  = (size_t)BB * sizeof(float);         // 16 KB

    bf16_t* Xhi     = (bf16_t*)(ws);
    bf16_t* Xlo     = (bf16_t*)(ws + XB);
    float*  norms   = (float*)(ws + 2 * XB);
    float*  possim  = (float*)(ws + 2 * XB + NORMB);
    float*  partial = (float*)(ws + 2 * XB + NORMB + POSB);  // 64*4096 floats

    prep_kernel<<<(2 * BB) / 8, 256, 0, stream>>>(X, Xhi, Xlo, norms);
    positives_kernel<<<BB / 8, 256, 0, stream>>>(X, norms, possim);
    gram_exp_kernel<<<dim3(32, 32), 256, SMEM_BYTES, stream>>>(Xhi, Xlo, norms, partial);
    finalize_kernel<<<1, 256, 0, stream>>>(partial, possim, out);
}
